// EncoderSimilarity_31164282700672
// MI455X (gfx1250) — matrix operand order repaired and checked by reading, not compiled, not run on hardware
//
#include <hip/hip_runtime.h>
#include <math.h>

// EncoderSimilarity (block-cosine sims) for MI455X / gfx1250, wave32.
//
// sims[a,b] = sum_t max_v <img_blk[a,v], cap_blk[b,t]>  for block sizes 128/256
// (blocks l2-normalized; cap globally l2-normalized first -> scalar factors).
// Raw 128-chunk dot products U[p,q] are computed on the bf16 matrix pipe with
// split precision (x = hi + lo, U ~= hi*hi + hi*lo + lo*hi, ~1e-5 rel error),
// using V_WMMA_F32_16X16X32_BF16 (16384 FLOP/instr vs 2048 for f32 WMMA).
// 256-chunk dots are U[2v',2t'] + U[2v'+1,2t'+1] (free reuse); all block
// normalizations are applied as positive scalar factors in the epilogue.

typedef float v8f __attribute__((ext_vector_type(8)));
typedef __bf16 bf16_t;
typedef bf16_t v4bf  __attribute__((ext_vector_type(4)));
typedef bf16_t v8bf  __attribute__((ext_vector_type(8)));
typedef bf16_t v16bf __attribute__((ext_vector_type(16)));

#define ES_EPS 1e-8f
#define NROW 2048
#define NDIM 1024

// A-plane LDS stride (bf16 units): 264 -> row-to-row spread of 4 banks.
#define SA 264
// B-plane LDS stride (bf16 units): 72 -> row-to-row spread of 36 banks.
#define SB 72

__device__ __forceinline__ v16bf cat8(v8bf a, v8bf b) {
  return __builtin_shufflevector(a, b, 0, 1, 2, 3, 4, 5, 6, 7, 8, 9, 10, 11,
                                 12, 13, 14, 15);
}

// Split fp32 quad into bf16 hi/lo planes: hi = bf16(x), lo = bf16(x - hi).
__device__ __forceinline__ void split4(float4 v, v4bf& h, v4bf& l) {
  float xs[4] = {v.x, v.y, v.z, v.w};
#pragma unroll
  for (int i = 0; i < 4; ++i) {
    bf16_t hh = (bf16_t)xs[i];
    h[i] = hh;
    l[i] = (bf16_t)(xs[i] - (float)hh);  // exact residual, then RNE to bf16
  }
}

// ---------------------------------------------------------------------------
// Kernel 1: per-row inverse block-norm scales (exact fp32).
//   img: scale = 1/(sqrt(ss_block) + EPS)
//   cap: global norm g = sqrt(tot)+EPS folds in: scale = 1/(sqrt(ss_block)+g*EPS)
// One wave32 per row; 8 waves per block.
// ---------------------------------------------------------------------------
__global__ __launch_bounds__(256) void es_scales_kernel(
    const float* __restrict__ X, float* __restrict__ S128,
    float* __restrict__ S256, int isCap) {
  const int row  = (blockIdx.x << 3) + (threadIdx.x >> 5);
  const int lane = threadIdx.x & 31;
  const float* xr = X + row * NDIM;

  float ss[8];
#pragma unroll
  for (int c = 0; c < 8; ++c) {
    float4 v = *(const float4*)(xr + c * 128 + lane * 4);
    float s = v.x * v.x + v.y * v.y + v.z * v.z + v.w * v.w;
#pragma unroll
    for (int off = 16; off >= 1; off >>= 1) s += __shfl_xor(s, off, 32);
    ss[c] = s;
  }
  if (lane == 0) {
    float tot = 0.f;
#pragma unroll
    for (int c = 0; c < 8; ++c) tot += ss[c];
    const float e = isCap ? (sqrtf(tot) + ES_EPS) * ES_EPS : ES_EPS;
#pragma unroll
    for (int q = 0; q < 8; ++q) S128[row * 8 + q] = 1.0f / (sqrtf(ss[q]) + e);
#pragma unroll
    for (int q = 0; q < 4; ++q)
      S256[row * 4 + q] = 1.0f / (sqrtf(ss[2 * q] + ss[2 * q + 1]) + e);
  }
}

// ---------------------------------------------------------------------------
// Kernel 2: fused split-bf16 block-cosine GEMM.
// Block = 256 threads = 8 waves; block tile = 32(a) x 64(b); each wave owns a
// 16x16 output tile. Loop over caption-chunk pairs Q (q = 2Q,2Q+1) holding
// acc[p][j] (8 img chunks x 2 cap chunks, 128 f32 VGPRs). K-window = 32 cols
// per chunk per stage; 6 bf16 WMMAs (hi*hi, hi*lo, lo*hi for j=0,1) per p.
//
// bf16 operand k order used here (contiguous, not the ISA table order): A(16x32): lane L holds
// 16 contiguous K at 16*(L>>4) of row L&15.  B(32x16): lane L holds
// 16 contiguous K at 16*(L>>4) of column (cap row) L&15.
// ---------------------------------------------------------------------------
__global__ __launch_bounds__(256, 1) void es_main_kernel(
    const float* __restrict__ img, const float* __restrict__ cap,
    const float* __restrict__ iS128, const float* __restrict__ iS256,
    const float* __restrict__ cS128, const float* __restrict__ cS256,
    float* __restrict__ out) {
  __shared__ bf16_t lAhi[32 * SA];  // 32 rows x (8 chunks x 32 K-cols)
  __shared__ bf16_t lAlo[32 * SA];
  __shared__ bf16_t lBhi[64 * SB];  // 64 rows x (2 chunks x 32 K-cols)
  __shared__ bf16_t lBlo[64 * SB];
  __shared__ float sI[32 * 12];     // [0..7]=128-blk scales, [8..11]=256-blk
  __shared__ float sC[64 * 12];

  const int tid   = threadIdx.x;
  const int lane  = tid & 31;
  const int wave  = tid >> 5;
  const int aBase = blockIdx.y << 5;
  const int bBase = blockIdx.x << 6;
  const int aOff  = (wave >> 2) << 4;  // 0 or 16
  const int bOff  = (wave & 3) << 4;   // 0..48
  const int frow  = lane & 15;
  const int hf    = lane >> 4;

  for (int i = tid; i < 384; i += 256) {
    const int row = i / 12, j = i % 12;
    sI[i] = (j < 8) ? iS128[(aBase + row) * 8 + j]
                    : iS256[(aBase + row) * 4 + (j - 8)];
  }
  for (int i = tid; i < 768; i += 256) {
    const int row = i / 12, j = i % 12;
    sC[i] = (j < 8) ? cS128[(bBase + row) * 8 + j]
                    : cS256[(bBase + row) * 4 + (j - 8)];
  }

  const v8f vzero = {};
  v8f sims = vzero;

  for (int Q = 0; Q < 4; ++Q) {  // caption-chunk pair (q = 2Q, 2Q+1)
    v8f acc[8][2];
#pragma unroll
    for (int p = 0; p < 8; ++p) { acc[p][0] = vzero; acc[p][1] = vzero; }

    for (int w = 0; w < 4; ++w) {  // 32-wide K-window within each 128-chunk
      __syncthreads();
      // Stage A: img rows aBase..+31, cols {128p + 32w + [0,32)} for all p,
      // split to bf16 hi/lo planes. 2048 float4s / 256 threads.
#pragma unroll
      for (int i = 0; i < 8; ++i) {
        const int idx = tid + (i << 8);
        const int row = idx >> 6, s = idx & 63, p = s >> 3, f = s & 7;
        const float4 v = *(const float4*)(img + (aBase + row) * NDIM +
                                          p * 128 + (w << 5) + (f << 2));
        v4bf h, l;
        split4(v, h, l);
        const int o = row * SA + (p << 5) + (f << 2);
        *(v4bf*)&lAhi[o] = h;
        *(v4bf*)&lAlo[o] = l;
      }
      // Stage B: cap rows bBase..+63, cols {128(2Q+j) + 32w + [0,32)}, j=0,1.
#pragma unroll
      for (int i = 0; i < 4; ++i) {
        const int idx = tid + (i << 8);
        const int row = idx >> 4, s = idx & 15, j = s >> 3, f = s & 7;
        const float4 v = *(const float4*)(cap + (bBase + row) * NDIM +
                                          (2 * Q + j) * 128 + (w << 5) +
                                          (f << 2));
        v4bf h, l;
        split4(v, h, l);
        const int o = row * SB + (j << 5) + (f << 2);
        *(v4bf*)&lBhi[o] = h;
        *(v4bf*)&lBlo[o] = l;
      }
      __syncthreads();

      // B fragments: 16 contiguous K at 16*hf inside each 32-col window.
      const int brow = (bOff + frow) * SB;
      const int bk = (hf << 4);
      const v16bf bh0 = cat8(*(const v8bf*)&lBhi[brow + bk],
                             *(const v8bf*)&lBhi[brow + bk + 8]);
      const v16bf bl0 = cat8(*(const v8bf*)&lBlo[brow + bk],
                             *(const v8bf*)&lBlo[brow + bk + 8]);
      const v16bf bh1 = cat8(*(const v8bf*)&lBhi[brow + 32 + bk],
                             *(const v8bf*)&lBhi[brow + 32 + bk + 8]);
      const v16bf bl1 = cat8(*(const v8bf*)&lBlo[brow + 32 + bk],
                             *(const v8bf*)&lBlo[brow + 32 + bk + 8]);

      const int arow = (aOff + frow) * SA;
      const int ak = (hf << 4);
#pragma unroll
      for (int p = 0; p < 8; ++p) {
        const int base = arow + (p << 5);
        const v16bf ah = cat8(*(const v8bf*)&lAhi[base + ak],
                              *(const v8bf*)&lAhi[base + 8 + ak]);
        const v16bf al = cat8(*(const v8bf*)&lAlo[base + ak],
                              *(const v8bf*)&lAlo[base + 8 + ak]);
        acc[p][0] = __builtin_amdgcn_wmma_f32_16x16x32_bf16(
            false, ah, false, bh0, (short)0, acc[p][0], false, false);
        acc[p][0] = __builtin_amdgcn_wmma_f32_16x16x32_bf16(
            false, ah, false, bl0, (short)0, acc[p][0], false, false);
        acc[p][0] = __builtin_amdgcn_wmma_f32_16x16x32_bf16(
            false, al, false, bh0, (short)0, acc[p][0], false, false);
        acc[p][1] = __builtin_amdgcn_wmma_f32_16x16x32_bf16(
            false, ah, false, bh1, (short)0, acc[p][1], false, false);
        acc[p][1] = __builtin_amdgcn_wmma_f32_16x16x32_bf16(
            false, ah, false, bl1, (short)0, acc[p][1], false, false);
        acc[p][1] = __builtin_amdgcn_wmma_f32_16x16x32_bf16(
            false, al, false, bh1, (short)0, acc[p][1], false, false);
      }
    }

    // Fused epilogue for q=2Q,2Q+1 (128-blk) and t'=Q (256-blk).
    // D layout: lane holds column n=frow; VGPR r holds row m = r + 8*hf.
    const float sc0 = sC[(bOff + frow) * 12 + 2 * Q];
    const float sc1 = sC[(bOff + frow) * 12 + 2 * Q + 1];
    const float sc2 = sC[(bOff + frow) * 12 + 8 + Q];
#pragma unroll
    for (int r = 0; r < 8; ++r) {
      const int m = aOff + (hf << 3) + r;
      float m0 = -3.402823466e38f, m1 = -3.402823466e38f,
            m2 = -3.402823466e38f;
#pragma unroll
      for (int v = 0; v < 4; ++v) {
        const float u00 = acc[2 * v][0][r];
        const float u01 = acc[2 * v][1][r];
        const float u10 = acc[2 * v + 1][0][r];
        const float u11 = acc[2 * v + 1][1][r];
        const float sA = sI[m * 12 + 2 * v];
        const float sB = sI[m * 12 + 2 * v + 1];
        m0 = fmaxf(m0, fmaxf(u00 * sA, u10 * sB));         // max_v, t=2Q
        m1 = fmaxf(m1, fmaxf(u01 * sA, u11 * sB));         // max_v, t=2Q+1
        m2 = fmaxf(m2, (u00 + u11) * sI[m * 12 + 8 + v]);  // 256-blk, t'=Q
      }
      sims[r] += m0 * sc0 + m1 * sc1 + m2 * sc2;  // cap scales > 0: hoisted
    }
  }

#pragma unroll
  for (int r = 0; r < 8; ++r) {
    const int a = aBase + aOff + (hf << 3) + r;
    const int b = bBase + bOff + frow;
    out[a * NROW + b] = sims[r];
  }
}

// ---------------------------------------------------------------------------
extern "C" void kernel_launch(void* const* d_in, const int* in_sizes, int n_in,
                              void* d_out, int out_size, void* d_ws,
                              size_t ws_size, hipStream_t stream) {
  (void)in_sizes; (void)n_in; (void)out_size; (void)ws_size;
  const float* img = (const float*)d_in[0];
  const float* cap = (const float*)d_in[1];
  float* out = (float*)d_out;

  float* ws    = (float*)d_ws;      // 192 KB of scale tables
  float* iS128 = ws;                // 2048*8
  float* iS256 = ws + 2048 * 8;     // 2048*4
  float* cS128 = iS256 + 2048 * 4;  // 2048*8
  float* cS256 = cS128 + 2048 * 8;  // 2048*4

  es_scales_kernel<<<NROW / 8, 256, 0, stream>>>(img, iS128, iS256, 0);
  es_scales_kernel<<<NROW / 8, 256, 0, stream>>>(cap, cS128, cS256, 1);
  es_main_kernel<<<dim3(NROW / 64, NROW / 32), 256, 0, stream>>>(
      img, cap, iS128, iS256, cS128, cS256, out);
}
